// TimeMix_10694468567121
// MI455X (gfx1250) — compile-verified, matrix operand order checked by reading, not run on hardware
//
#include <hip/hip_runtime.h>
#include <hip/hip_bf16.h>

typedef __attribute__((ext_vector_type(16))) __bf16 v16bf;
typedef __attribute__((ext_vector_type(8)))  float  v8f;

__device__ __forceinline__ __bf16 f2bf(float f) { return (__bf16)f; }

// Async global->LDS DMA (CDNA5, tracked by ASYNCcnt). VDST operand carries the
// LDS byte address (low 32 bits of the generic shared pointer).
__device__ __forceinline__ void async_b128(unsigned lds, const void* g) {
  asm volatile("global_load_async_to_lds_b128 %0, %1, off"
               :: "v"(lds), "v"(g) : "memory");
}
__device__ __forceinline__ void wait_async0() {
  asm volatile("s_wait_asynccnt 0x0" ::: "memory");
}
__device__ __forceinline__ unsigned lds_off(const void* p) {
  return (unsigned)(uintptr_t)p;
}

// ---------------------------------------------------------------------------
// bf16 WMMA GEMM:  C[M,N] (f32) = A[M,K] (bf16, row-major, lda)
//                               x Bt[N,K]^T (bf16, row-major, ldb = K stride)
// Block tile 128x128, 8 waves of 32x64, K-step 32 via v_wmma_f32_16x16x32_bf16.
// Double-buffered LDS staged by GLOBAL_LOAD_ASYNC_TO_LDS_B128.
// ---------------------------------------------------------------------------
#define BM 128
#define BN 128
#define BK 32
#define LDT 40   // LDS row pitch in bf16 (80B: 16B-aligned, conflict-free)

__global__ __launch_bounds__(256) void gemm_bf16_kernel(
    const __bf16* __restrict__ A, int lda,
    const __bf16* __restrict__ Bt, int ldb,
    float* __restrict__ C, int ldc,
    int M, int N, int K)
{
  __shared__ __attribute__((aligned(16))) __bf16 As[2][BM * LDT];
  __shared__ __attribute__((aligned(16))) __bf16 Bs[2][BN * LDT];

  const int tid  = threadIdx.x;
  const int lane = tid & 31;
  const int wid  = tid >> 5;          // 0..7
  const int wm   = (wid >> 1) * 32;   // wave row offset in block tile
  const int wn   = (wid & 1) * 64;    // wave col offset in block tile
  const int bm0  = blockIdx.y * BM;
  const int bn0  = blockIdx.x * BN;
  const int hi   = lane >> 4;         // lane half (selects K sub-range)
  const int l15  = lane & 15;         // M (A) / N (B,C) index within tile

  // staging coordinates for this thread (2 x 16B chunks per matrix)
  const int idx0 = tid * 8;
  const int row0 = idx0 >> 5, col0 = idx0 & 31;
  const int idx1 = (tid + 256) * 8;
  const int row1 = idx1 >> 5, col1 = idx1 & 31;

  v8f acc[2][4];
#pragma unroll
  for (int i = 0; i < 2; ++i)
#pragma unroll
    for (int j = 0; j < 4; ++j)
#pragma unroll
      for (int e = 0; e < 8; ++e) acc[i][j][e] = 0.0f;

  // issue async stage of K-slab `k0` into buffer `buf`
  auto stage = [&](int k0, int buf) {
    // A tile: always in-bounds (M multiple of 128, K multiple of 32)
    async_b128(lds_off(&As[buf][row0 * LDT + col0]),
               A + (size_t)(bm0 + row0) * lda + k0 + col0);
    async_b128(lds_off(&As[buf][row1 * LDT + col1]),
               A + (size_t)(bm0 + row1) * lda + k0 + col1);
    // B tile: zero-fill rows beyond N via plain LDS store
    int n0r = bn0 + row0;
    if (n0r < N)
      async_b128(lds_off(&Bs[buf][row0 * LDT + col0]),
                 Bt + (size_t)n0r * ldb + k0 + col0);
    else
      *(uint4*)(&Bs[buf][row0 * LDT + col0]) = make_uint4(0u, 0u, 0u, 0u);
    int n1r = bn0 + row1;
    if (n1r < N)
      async_b128(lds_off(&Bs[buf][row1 * LDT + col1]),
                 Bt + (size_t)n1r * ldb + k0 + col1);
    else
      *(uint4*)(&Bs[buf][row1 * LDT + col1]) = make_uint4(0u, 0u, 0u, 0u);
  };

  const int nk = K / BK;
  stage(0, 0);

  for (int kk = 0; kk < nk; ++kk) {
    // make this wave's DMA visible, then sync all waves
    wait_async0();
    __syncthreads();

    const int cur = kk & 1;
    if (kk + 1 < nk) {
      stage((kk + 1) * BK, cur ^ 1);   // overlap next slab with compute
      __builtin_prefetch(A + (size_t)(bm0 + (tid & 127)) * lda + (kk + 1) * BK, 0, 1);
    }

    const __bf16* Abase = &As[cur][0];
    const __bf16* Bbase = &Bs[cur][0];

    // ---- load fragments per the CDNA5 16-bit A/B VGPR layout ----
    v16bf af[2];
#pragma unroll
    for (int i = 0; i < 2; ++i) {
      const __bf16* base = Abase + (wm + 16 * i + l15) * LDT;
#pragma unroll
      for (int v = 0; v < 8; ++v) {
        int kb = ((v >> 2) << 4) + (hi << 3) + ((v & 3) << 1);
        af[i][2 * v]     = base[kb];
        af[i][2 * v + 1] = base[kb + 1];
      }
    }
    v16bf bfm[4];
#pragma unroll
    for (int j = 0; j < 4; ++j) {
      const __bf16* base = Bbase + (wn + 16 * j + l15) * LDT;
#pragma unroll
      for (int v = 0; v < 8; ++v) {
        int kb = ((v >> 2) << 4) + (hi << 3) + ((v & 3) << 1);
        bfm[j][2 * v]     = base[kb];
        bfm[j][2 * v + 1] = base[kb + 1];
      }
    }

#pragma unroll
    for (int i = 0; i < 2; ++i)
#pragma unroll
      for (int j = 0; j < 4; ++j)
        acc[i][j] = __builtin_amdgcn_wmma_f32_16x16x32_bf16(
            false, af[i], false, bfm[j], (short)0, acc[i][j], false, false);
  }

  // ---- store: C layout VGPR r -> M = r + 8*hi, N = lane&15 ----
#pragma unroll
  for (int i = 0; i < 2; ++i) {
    int gm = bm0 + wm + 16 * i + hi * 8;
#pragma unroll
    for (int j = 0; j < 4; ++j) {
      int gn = bn0 + wn + 16 * j + l15;
      if (gn < N) {
#pragma unroll
        for (int r = 0; r < 8; ++r)
          C[(size_t)(gm + r) * ldc + gn] = acc[i][j][r];
      }
    }
  }
}

// ---------------------------------------------------------------------------
// Elementwise / prep kernels
// ---------------------------------------------------------------------------
__global__ void convert_nt_kernel(const float* __restrict__ s, __bf16* __restrict__ d, int n) {
  int i = blockIdx.x * blockDim.x + threadIdx.x;
  if (i < n) d[i] = f2bf(s[i]);
}

// src [R][Cc] -> dst [Cc][R]  (transpose to [N,K] row-major, bf16)
__global__ void convert_t_kernel(const float* __restrict__ s, __bf16* __restrict__ d, int R, int Cc) {
  int i = blockIdx.x * blockDim.x + threadIdx.x;
  if (i < R * Cc) {
    int r = i / Cc, c = i % Cc;
    d[(size_t)c * R + r] = f2bf(s[i]);
  }
}

__global__ void shift_kernel(const float* __restrict__ x, const float* __restrict__ xmiu,
                             float* __restrict__ dx, __bf16* __restrict__ xm,
                             int T, int Cc, int n) {
  int i = blockIdx.x * blockDim.x + threadIdx.x;
  if (i >= n) return;
  int c = i % Cc;
  int t = (i / Cc) % T;
  float xv = x[i];
  float xl = (t == 0) ? 0.0f : x[i - Cc];
  float d  = xl - xv;
  dx[i] = d;
  xm[i] = f2bf(xv + d * xmiu[c]);
}

__global__ void tanh_bf16_kernel(const float* __restrict__ s, __bf16* __restrict__ d, int n) {
  int i = blockIdx.x * blockDim.x + threadIdx.x;
  if (i < n) d[i] = f2bf(tanhf(s[i]));
}

__global__ void cast_bf16_kernel(const float* __restrict__ s, __bf16* __restrict__ d, int n) {
  int i = blockIdx.x * blockDim.x + threadIdx.x;
  if (i < n) d[i] = f2bf(s[i]);
}

// xd = bf16(x + dx*(lambda[c] + miu))
__global__ void xd_kernel(const float* __restrict__ x, const float* __restrict__ dx,
                          const float* __restrict__ lam, const float* __restrict__ miu,
                          __bf16* __restrict__ xd, int Cc, int n) {
  int i = blockIdx.x * blockDim.x + threadIdx.x;
  if (i >= n) return;
  int c = i % Cc;
  xd[i] = f2bf(x[i] + dx[i] * (lam[c] + miu[i]));
}

// w = -softplus(-(w_miu + src)) - 0.5   (numerically stable)
__global__ void wgate_kernel(const float* __restrict__ src, const float* __restrict__ wmiu,
                             float* __restrict__ w, int Cc, int n) {
  int i = blockIdx.x * blockDim.x + threadIdx.x;
  if (i >= n) return;
  int c = i % Cc;
  float u  = wmiu[c] + src[i];
  float sp = (u > 0.0f) ? log1pf(expf(-u)) : (log1pf(expf(u)) - u);
  w[i] = -sp - 0.5f;
}

// a_gate = 2*sigmoid(a_time + src)
__global__ void agate_kernel(const float* __restrict__ src, const float* __restrict__ atime,
                             float* __restrict__ a, int Cc, int n) {
  int i = blockIdx.x * blockDim.x + threadIdx.x;
  if (i >= n) return;
  int c = i % Cc;
  float u = atime[c] + src[i];
  a[i] = 2.0f / (1.0f + expf(-u));
}

// k = kraw + kl, then row L2-normalize over Cc columns
__global__ __launch_bounds__(256) void knorm_kernel(const float* __restrict__ kraw,
                                                    const float* __restrict__ kl,
                                                    float* __restrict__ kout, int Cc) {
  __shared__ float red[256];
  int row = blockIdx.x;
  int tid = threadIdx.x;
  float ss = 0.0f;
  for (int c = tid; c < Cc; c += 256) {
    float kv = kraw[(size_t)row * Cc + c] + kl[(size_t)row * Cc + c];
    kout[(size_t)row * Cc + c] = kv;
    ss += kv * kv;
  }
  red[tid] = ss;
  __syncthreads();
  for (int s = 128; s > 0; s >>= 1) {
    if (tid < s) red[tid] += red[tid + s];
    __syncthreads();
  }
  float scale = 1.0f / fmaxf(sqrtf(red[0]), 1e-12f);
  for (int c = tid; c < Cc; c += 256) kout[(size_t)row * Cc + c] *= scale;
}

// ---------------------------------------------------------------------------
// wkv7 scan: one block per (b,h); thread i owns row i of the 64x64 state.
// S = S*diag(exp(w)) + (S a) b^T + v kk^T ; y = S r
// a = -k, b = k*a_gate, kk = k*exp(min(w/2,0)) computed inline.
// ---------------------------------------------------------------------------
__global__ __launch_bounds__(64) void wkv7_kernel(
    const float* __restrict__ r, const float* __restrict__ w,
    const float* __restrict__ k, const float* __restrict__ v,
    const float* __restrict__ ag, float* __restrict__ y,
    int T, int H) {
  const int C = H * 64;
  int bh = blockIdx.x;
  int b  = bh / H;
  int h  = bh % H;
  int i  = threadIdx.x;  // value dim

  __shared__ float rL[64], dL[64], kL[64], aL[64], bL[64];
  float S[64];
#pragma unroll
  for (int j = 0; j < 64; ++j) S[j] = 0.0f;

  size_t base = (size_t)(b * T) * C + h * 64;
  for (int t = 0; t < T; ++t, base += C) {
    float wv = w[base + i];
    float kv = k[base + i];
    rL[i] = r[base + i];
    dL[i] = expf(wv);
    kL[i] = kv * expf(fminf(0.5f * wv, 0.0f));
    aL[i] = -kv;
    bL[i] = kv * ag[base + i];
    float vi = v[base + i];
    __syncthreads();

    float sa = 0.0f;
#pragma unroll
    for (int j = 0; j < 64; ++j) sa += S[j] * aL[j];
    float yv = 0.0f;
#pragma unroll
    for (int j = 0; j < 64; ++j) {
      S[j] = S[j] * dL[j] + sa * bL[j] + vi * kL[j];
      yv += S[j] * rL[j];
    }
    y[base + i] = yv;
    __syncthreads();
  }
}

// ---------------------------------------------------------------------------
// GroupNorm over 64-dim heads + bonus + gate multiply; one block per (row,head)
// z = bf16( (GN(y)*gamma+beta + ((r*k*x_time).sum) * v) * g )
// ---------------------------------------------------------------------------
__device__ __forceinline__ float blockReduce64(float val, float* red, int n) {
  red[n] = val;
  __syncthreads();
#pragma unroll
  for (int s = 32; s > 0; s >>= 1) {
    if (n < s) red[n] += red[n + s];
    __syncthreads();
  }
  float out = red[0];
  __syncthreads();
  return out;
}

__global__ __launch_bounds__(64) void gnb_kernel(
    const float* __restrict__ y, const float* __restrict__ r,
    const float* __restrict__ k, const float* __restrict__ v,
    const float* __restrict__ g, const float* __restrict__ gamma,
    const float* __restrict__ beta, const float* __restrict__ xt,
    __bf16* __restrict__ z, int H) {
  const int C = H * 64;
  __shared__ float red[64];
  int mh = blockIdx.x;
  int m  = mh / H;
  int h  = mh % H;
  int n  = threadIdx.x;
  int c  = h * 64 + n;
  size_t idx = (size_t)m * C + c;

  float yv = y[idx];
  float mean  = blockReduce64(yv, red, n) * (1.0f / 64.0f);
  float sq    = blockReduce64(yv * yv, red, n) * (1.0f / 64.0f);
  float var   = sq - mean * mean;
  float eps   = 1e-5f * (float)H;
  float yn    = (yv - mean) * rsqrtf(var + eps);
  float yout  = yn * gamma[c] + beta[c];

  float dot   = blockReduce64(r[idx] * k[idx] * xt[h * 64 + n], red, n);
  float bonus = dot * v[idx];
  z[idx] = f2bf((yout + bonus) * g[idx]);
}

// ---------------------------------------------------------------------------
// Launch orchestration
// ---------------------------------------------------------------------------
extern "C" void kernel_launch(void* const* d_in, const int* in_sizes, int n_in,
                              void* d_out, int out_size, void* d_ws, size_t ws_size,
                              hipStream_t stream) {
  (void)in_sizes; (void)n_in; (void)out_size; (void)ws_size;
  const int Bn = 4, T = 1024, Cc = 1024, H = 16;
  const int M = Bn * T;  // 4096

  const float* x       = (const float*)d_in[0];
  const float* lambdas = (const float*)d_in[1];
  const float* k_A     = (const float*)d_in[2];
  const float* k_B     = (const float*)d_in[3];
  const float* g_A     = (const float*)d_in[4];
  const float* g_B     = (const float*)d_in[5];
  const float* w_A     = (const float*)d_in[6];
  const float* w_B     = (const float*)d_in[7];
  const float* w_miu   = (const float*)d_in[8];
  const float* a_A     = (const float*)d_in[9];
  const float* a_B     = (const float*)d_in[10];
  const float* a_time  = (const float*)d_in[11];
  const float* x_A     = (const float*)d_in[12];
  const float* x_B     = (const float*)d_in[13];
  const float* x_miu   = (const float*)d_in[14];
  const float* x_time  = (const float*)d_in[15];
  const float* gng     = (const float*)d_in[16];
  const float* gnb     = (const float*)d_in[17];
  const float* rW      = (const float*)d_in[18];
  const float* kW      = (const float*)d_in[19];
  const float* vW      = (const float*)d_in[20];
  const float* oW      = (const float*)d_in[21];

  char* ws = (char*)d_ws;
  size_t off = 0;
  auto A256 = [&](size_t bytes) { size_t o = off; off += (bytes + 255) & ~(size_t)255; return o; };
  auto BF   = [&](size_t elems) { return (__bf16*)(ws + A256(elems * 2)); };
  auto F32  = [&](size_t elems) { return (float*)(ws + A256(elems * 4)); };

  // bf16 weights ([N,K] row-major for "x @ W^T" fragments)
  __bf16* rWb = BF((size_t)Cc * Cc);
  __bf16* kWb = BF((size_t)Cc * Cc);
  __bf16* vWb = BF((size_t)Cc * Cc);
  __bf16* oWb = BF((size_t)Cc * Cc);
  __bf16* xAt = BF((size_t)192 * Cc);     // [192][1024]
  __bf16* xBt = BF((size_t)6 * Cc * 32);  // 6 x [1024][32]
  __bf16* kAt = BF((size_t)64 * Cc);
  __bf16* kBt = BF((size_t)Cc * 64);
  __bf16* gAt = BF((size_t)128 * Cc);
  __bf16* gBt = BF((size_t)Cc * 128);
  __bf16* wAt = BF((size_t)64 * Cc);
  __bf16* wBt = BF((size_t)Cc * 64);
  __bf16* aAt = BF((size_t)64 * Cc);
  __bf16* aBt = BF((size_t)Cc * 64);

  // activations
  float*  dxf    = F32((size_t)M * Cc);
  __bf16* xmb    = BF((size_t)M * Cc);
  float*  mixraw = F32((size_t)M * 192);
  __bf16* mixb   = BF((size_t)M * 192);
  float*  T0     = F32((size_t)M * Cc);
  float*  T1     = F32((size_t)M * Cc);
  float*  lmidf  = F32((size_t)M * 128);
  __bf16* lmidb  = BF((size_t)M * 128);
  __bf16* xdb    = BF((size_t)M * Cc);
  float*  rbuf   = F32((size_t)M * Cc);
  float*  wbuf   = F32((size_t)M * Cc);
  float*  kbuf   = F32((size_t)M * Cc);
  float*  vbuf   = F32((size_t)M * Cc);
  float*  gbuf   = F32((size_t)M * Cc);
  float*  abuf   = F32((size_t)M * Cc);
  float*  ybuf   = F32((size_t)M * Cc);
  __bf16* zb     = BF((size_t)M * Cc);

  auto blk = [](int n) { return (n + 255) / 256; };
  auto cvt_nt = [&](const float* s, __bf16* d, int n) {
    convert_nt_kernel<<<blk(n), 256, 0, stream>>>(s, d, n);
  };
  auto cvt_t = [&](const float* s, __bf16* d, int R, int Ccols) {
    int n = R * Ccols;
    convert_t_kernel<<<blk(n), 256, 0, stream>>>(s, d, R, Ccols);
  };
  auto gemm = [&](const __bf16* Ab, int lda, const __bf16* Btb, int ldb,
                  float* Cf, int ldc, int Mm, int Nn, int Kk) {
    dim3 grid((Nn + BN - 1) / BN, Mm / BM);
    gemm_bf16_kernel<<<grid, 256, 0, stream>>>(Ab, lda, Btb, ldb, Cf, ldc, Mm, Nn, Kk);
  };

  // ---- weight prep ----
  cvt_nt(rW, rWb, Cc * Cc);
  cvt_nt(kW, kWb, Cc * Cc);
  cvt_nt(vW, vWb, Cc * Cc);
  cvt_nt(oW, oWb, Cc * Cc);
  cvt_t(x_A, xAt, Cc, 192);
  for (int g6 = 0; g6 < 6; ++g6)
    cvt_t(x_B + (size_t)g6 * 32 * Cc, xBt + (size_t)g6 * Cc * 32, 32, Cc);
  cvt_t(k_A, kAt, Cc, 64);
  cvt_t(k_B, kBt, 64, Cc);
  cvt_t(g_A, gAt, Cc, 128);
  cvt_t(g_B, gBt, 128, Cc);
  cvt_t(w_A, wAt, Cc, 64);
  cvt_t(w_B, wBt, 64, Cc);
  cvt_t(a_A, aAt, Cc, 64);
  cvt_t(a_B, aBt, 64, Cc);

  const int nMC = M * Cc;

  // ---- time shift + ddlerp mix ----
  shift_kernel<<<blk(nMC), 256, 0, stream>>>(x, x_miu, dxf, xmb, T, Cc, nMC);
  gemm(xmb, Cc, xAt, Cc, mixraw, 192, M, 192, Cc);
  tanh_bf16_kernel<<<blk(M * 192), 256, 0, stream>>>(mixraw, mixb, M * 192);

  // gates: 0=r, 1=w, 2=k, 3=v, 4=a, 5=g
  for (int g6 = 0; g6 < 6; ++g6) {
    gemm(mixb + g6 * 32, 192, xBt + (size_t)g6 * Cc * 32, 32, T0, Cc, M, Cc, 32);
    xd_kernel<<<blk(nMC), 256, 0, stream>>>(x, dxf, lambdas + g6 * Cc, T0, xdb, Cc, nMC);

    if (g6 == 0) {
      gemm(xdb, Cc, rWb, Cc, rbuf, Cc, M, Cc, Cc);
    } else if (g6 == 1) {
      gemm(xdb, Cc, wAt, Cc, lmidf, 64, M, 64, Cc);
      tanh_bf16_kernel<<<blk(M * 64), 256, 0, stream>>>(lmidf, lmidb, M * 64);
      gemm(lmidb, 64, wBt, 64, T1, Cc, M, Cc, 64);
      wgate_kernel<<<blk(nMC), 256, 0, stream>>>(T1, w_miu, wbuf, Cc, nMC);
    } else if (g6 == 2) {
      gemm(xdb, Cc, kWb, Cc, T1, Cc, M, Cc, Cc);
      gemm(xdb, Cc, kAt, Cc, lmidf, 64, M, 64, Cc);
      tanh_bf16_kernel<<<blk(M * 64), 256, 0, stream>>>(lmidf, lmidb, M * 64);
      gemm(lmidb, 64, kBt, 64, T0, Cc, M, Cc, 64);
      knorm_kernel<<<M, 256, 0, stream>>>(T1, T0, kbuf, Cc);
    } else if (g6 == 3) {
      gemm(xdb, Cc, vWb, Cc, vbuf, Cc, M, Cc, Cc);
    } else if (g6 == 4) {
      gemm(xdb, Cc, aAt, Cc, lmidf, 64, M, 64, Cc);
      cast_bf16_kernel<<<blk(M * 64), 256, 0, stream>>>(lmidf, lmidb, M * 64);
      gemm(lmidb, 64, aBt, 64, T1, Cc, M, Cc, 64);
      agate_kernel<<<blk(nMC), 256, 0, stream>>>(T1, a_time, abuf, Cc, nMC);
    } else {
      gemm(xdb, Cc, gAt, Cc, lmidf, 128, M, 128, Cc);
      tanh_bf16_kernel<<<blk(M * 128), 256, 0, stream>>>(lmidf, lmidb, M * 128);
      gemm(lmidb, 128, gBt, 128, gbuf, Cc, M, Cc, 128);
    }
  }

  // ---- sequential wkv7 scan ----
  wkv7_kernel<<<Bn * H, 64, 0, stream>>>(rbuf, wbuf, kbuf, vbuf, abuf, ybuf, T, H);

  // ---- groupnorm + bonus + gate ----
  gnb_kernel<<<M * H, 64, 0, stream>>>(ybuf, rbuf, kbuf, vbuf, gbuf, gng, gnb, x_time, zb, H);

  // ---- output projection ----
  gemm(zb, Cc, oWb, Cc, (float*)d_out, Cc, M, Cc, Cc);
}
